// TransformerBlock_62234076119821
// MI455X (gfx1250) — matrix operand order repaired and checked by reading, not compiled, not run on hardware
//
#include <hip/hip_runtime.h>
#include <hip/hip_bf16.h>

typedef __attribute__((ext_vector_type(16))) _Float16 v16h;
typedef __attribute__((ext_vector_type(8)))  float    v8f;
typedef _Float16 half_t;

constexpr int kB = 8, kS = 1024, kD = 1024, kH = 16, kHD = 64, kFF = 4096;
constexpr int kM = kB * kS;               // 8192 tokens

// CDNA5 async direct global->LDS copy (bypasses VGPRs, tracked by ASYNCcnt).
#define USE_ASYNC 1

static __device__ __forceinline__
void async_copy_b128(void* lds_dst, const void* gsrc) {
#if USE_ASYNC
  unsigned ldsa = (unsigned)(unsigned long long)lds_dst;  // low 32b = LDS offset
  unsigned long long ga = (unsigned long long)gsrc;
  asm volatile("global_load_async_to_lds_b128 %0, %1, off"
               :: "v"(ldsa), "v"(ga) : "memory");
#else
  *(uint4*)lds_dst = *(const uint4*)gsrc;
#endif
}

static __device__ __forceinline__ void async_wait() {
#if USE_ASYNC
  asm volatile("s_wait_asynccnt 0" ::: "memory");
#endif
}

// ---------------------------------------------------------------- utilities
__global__ __launch_bounds__(256)
void cvt_f16(const float* __restrict__ in, half_t* __restrict__ out, int n4) {
  int i = blockIdx.x * blockDim.x + threadIdx.x;
  if (i < n4) {
    float4 f = ((const float4*)in)[i];
    alignas(8) half_t h[4];
    h[0] = (half_t)f.x; h[1] = (half_t)f.y; h[2] = (half_t)f.z; h[3] = (half_t)f.w;
    *(uint2*)(out + (size_t)i * 4) = *(uint2*)h;
  }
}

static __device__ __forceinline__
float block_reduce(float val, float* red, int t) {
  red[t] = val; __syncthreads();
  #pragma unroll
  for (int off = 128; off > 0; off >>= 1) {
    if (t < off) red[t] += red[t + off];
    __syncthreads();
  }
  float r = red[0];
  __syncthreads();
  return r;
}

// Fused LayerNorm (optionally double: LN(LN(x,g1,b1),g2,b2)) -> f16 output
__global__ __launch_bounds__(256)
void ln_kernel(const float* __restrict__ x,
               const float* __restrict__ g1, const float* __restrict__ b1,
               const float* __restrict__ g2, const float* __restrict__ b2,
               half_t* __restrict__ out, int dbl) {
  __shared__ float red[256];
  const int t = threadIdx.x;
  const size_t row = blockIdx.x;
  const float* xr = x + row * kD;
  float v[4];
  #pragma unroll
  for (int i = 0; i < 4; ++i) v[i] = xr[t + i * 256];

  float mu = block_reduce(v[0] + v[1] + v[2] + v[3], red, t) * (1.0f / kD);
  float ss = 0.f;
  #pragma unroll
  for (int i = 0; i < 4; ++i) { float d = v[i] - mu; ss += d * d; }
  float r = rsqrtf(block_reduce(ss, red, t) * (1.0f / kD) + 1e-6f);

  float y[4];
  #pragma unroll
  for (int i = 0; i < 4; ++i)
    y[i] = (v[i] - mu) * r * g1[t + i * 256] + b1[t + i * 256];

  if (dbl) {
    float mu2 = block_reduce(y[0] + y[1] + y[2] + y[3], red, t) * (1.0f / kD);
    float ss2 = 0.f;
    #pragma unroll
    for (int i = 0; i < 4; ++i) { float d = y[i] - mu2; ss2 += d * d; }
    float r2 = rsqrtf(block_reduce(ss2, red, t) * (1.0f / kD) + 1e-6f);
    #pragma unroll
    for (int i = 0; i < 4; ++i)
      y[i] = (y[i] - mu2) * r2 * g2[t + i * 256] + b2[t + i * 256];
  }
  #pragma unroll
  for (int i = 0; i < 4; ++i) out[row * kD + t + i * 256] = (half_t)y[i];
}

// ---------------------------------------------------------------- WMMA GEMM
// C(MxN) = A(MxK,f16) @ W(KxN,f16) + bias, then per MODE:
//   0: store f16      1: exact GELU then store f16      2: +resid, store f32
// Block = 128 threads = 4 waves; block tile 128x64; wave tile 32x64; BK=32.
constexpr int BM = 128, BN = 64, BK = 32, LDA = 40, LDB = 40;

template <int MODE>
__global__ __launch_bounds__(128)
void gemm_wmma(const half_t* __restrict__ A, const half_t* __restrict__ W,
               const float* __restrict__ bias, const float* __restrict__ resid,
               float* __restrict__ outF, half_t* __restrict__ outH,
               int M, int N, int K) {
  __shared__ half_t As[BM * LDA];   // row-major, padded (stride 40 halves)
  __shared__ half_t Bs[BN * LDB];   // W transposed: Bs[n][k], padded
  const int tid  = threadIdx.x;
  const int wave = tid >> 5, lane = tid & 31;
  const int lr = lane & 15, hi = lane >> 4;
  const int m0 = blockIdx.y * BM;
  const int n0 = blockIdx.x * BN;

  v8f acc[2][4] = {};
  const int kTiles = K / BK;

  for (int kt = 0; kt < kTiles; ++kt) {
    // stage A tile async: thread t owns row t (32 halves = 4 x 16B, direct to LDS)
    {
      const half_t* src = A + (size_t)(m0 + tid) * K + kt * BK;
      half_t* dst = As + tid * LDA;
      #pragma unroll
      for (int c = 0; c < 4; ++c)
        async_copy_b128(dst + c * 8, src + c * 8);
      if (kt + 1 < kTiles)
        __builtin_prefetch(src + BK, 0, 1);
    }
    // stage W tile transposed: thread t: k = t/4, 16 cols starting (t%4)*16
    {
      int kk = tid >> 2, seg = (tid & 3) * 16;
      const uint4* src = (const uint4*)(W + (size_t)(kt * BK + kk) * N + n0 + seg);
      alignas(16) half_t tmp[16];
      *(uint4*)tmp = src[0]; *(uint4*)(tmp + 8) = src[1];
      #pragma unroll
      for (int j = 0; j < 16; ++j) Bs[(seg + j) * LDB + kk] = tmp[j];
    }
    async_wait();
    __syncthreads();

    // A fragments: lane holds row lr, K = {hi*8+j, 16+hi*8+j}
    v16h af[2];
    #pragma unroll
    for (int rs = 0; rs < 2; ++rs) {
      const uint4* p = (const uint4*)(As + (wave * 32 + rs * 16 + lr) * LDA);
      v16h a; ((uint4*)&a)[0] = p[hi]; ((uint4*)&a)[1] = p[2 + hi];
      af[rs] = a;
    }
    // B fragments: lane holds col lr, K = {hi*8+j, 16+hi*8+j}
    #pragma unroll
    for (int g = 0; g < 4; ++g) {
      const uint4* p = (const uint4*)(Bs + (g * 16 + lr) * LDB);
      v16h b; ((uint4*)&b)[0] = p[hi]; ((uint4*)&b)[1] = p[2 + hi];
      acc[0][g] = __builtin_amdgcn_wmma_f32_16x16x32_f16(
          false, af[0], false, b, (short)0, acc[0][g], false, false);
      acc[1][g] = __builtin_amdgcn_wmma_f32_16x16x32_f16(
          false, af[1], false, b, (short)0, acc[1][g], false, false);
    }
    __syncthreads();
  }

  // epilogue: C/D layout -> row m = j + 8*hi (VGPR j), col n = lr
  #pragma unroll
  for (int rs = 0; rs < 2; ++rs) {
    #pragma unroll
    for (int g = 0; g < 4; ++g) {
      const int col = n0 + g * 16 + lr;
      const float bv = bias[col];
      #pragma unroll
      for (int j = 0; j < 8; ++j) {
        const int row = m0 + wave * 32 + rs * 16 + hi * 8 + j;
        float val = acc[rs][g][j] + bv;
        size_t idx = (size_t)row * N + col;
        if constexpr (MODE == 0) {
          outH[idx] = (half_t)val;
        } else if constexpr (MODE == 1) {
          float ge = 0.5f * val * (1.0f + erff(val * 0.70710678118654752f));
          outH[idx] = (half_t)ge;
        } else {
          outF[idx] = resid[idx] + val;
        }
      }
    }
  }
}

// -------------------------------------------------- fused flash attention
// grid (S/64, H, B); 128 threads = 4 waves; each wave owns 16 query rows,
// iterates over key blocks of 32 (so P@V is a K=32 WMMA).
__global__ __launch_bounds__(128)
void attn_wmma(const half_t* __restrict__ q, const half_t* __restrict__ k,
               const half_t* __restrict__ v, half_t* __restrict__ ctx) {
  constexpr int LDK = 72, LDV = 40, LDP = 40;
  __shared__ half_t Ks[32 * LDK];      // K tile row-major (key, hd)  == B^T
  __shared__ half_t Vt[64 * LDV];      // V tile transposed (hd, key) == B^T
  __shared__ half_t Ps[4][16 * LDP];   // per-wave probs, C->A relayout
  const int tid = threadIdx.x;
  const int wave = tid >> 5, lane = tid & 31;
  const int lr = lane & 15, hi = lane >> 4;
  const int b = blockIdx.z, h = blockIdx.y;
  const int q0 = blockIdx.x * 64 + wave * 16;
  const size_t hoff = (size_t)h * kHD;

  // Q fragments (K = head dim 64 -> two k=32 fragments), A layout from global
  v16h aq[2];
  {
    const uint4* p = (const uint4*)(q + ((size_t)(b * kS + q0 + lr) * kD) + hoff);
    #pragma unroll
    for (int kk = 0; kk < 2; ++kk) {
      v16h a; ((uint4*)&a)[0] = p[kk * 4 + hi]; ((uint4*)&a)[1] = p[kk * 4 + 2 + hi];
      aq[kk] = a;
    }
  }

  float mrow[8], lrow[8];
  v8f O[4] = {};
  #pragma unroll
  for (int j = 0; j < 8; ++j) { mrow[j] = -1e30f; lrow[j] = 0.f; }

  for (int kb = 0; kb < kS / 32; ++kb) {
    const int key0 = kb * 32;
    // stage K async (row major tile is already B^T for Q.K^T); V transposed
    {
      int key = tid >> 2, seg = (tid & 3) * 16;
      const size_t tok = (size_t)(b * kS + key0 + key) * kD + hoff + seg;
      async_copy_b128(Ks + key * LDK + seg,     k + tok);
      async_copy_b128(Ks + key * LDK + seg + 8, k + tok + 8);
      const uint4* vs = (const uint4*)(v + tok);
      alignas(16) half_t tmp[16];
      *(uint4*)tmp = vs[0]; *(uint4*)(tmp + 8) = vs[1];
      #pragma unroll
      for (int j = 0; j < 16; ++j) Vt[(seg + j) * LDV + key] = tmp[j];
    }
    async_wait();
    __syncthreads();

    // scores: two 16x16 fragments (key groups), K-dim = 64 head dims
    v8f s[2] = {};
    #pragma unroll
    for (int kk = 0; kk < 2; ++kk) {
      #pragma unroll
      for (int grp = 0; grp < 2; ++grp) {
        const uint4* p = (const uint4*)(Ks + (grp * 16 + lr) * LDK + kk * 32);
        v16h bf; ((uint4*)&bf)[0] = p[hi]; ((uint4*)&bf)[1] = p[2 + hi];
        s[grp] = __builtin_amdgcn_wmma_f32_16x16x32_f16(
            false, aq[kk], false, bf, (short)0, s[grp], false, false);
      }
    }

    // online softmax; row m = j + 8*hi lives across 16 lanes of one half-wave
    float alpha[8];
    #pragma unroll
    for (int j = 0; j < 8; ++j) {
      float s0 = s[0][j] * 0.125f;           // 1/sqrt(64)
      float s1 = s[1][j] * 0.125f;
      float mx = fmaxf(s0, s1);
      #pragma unroll
      for (int d = 1; d < 16; d <<= 1) mx = fmaxf(mx, __shfl_xor(mx, d, 32));
      float mnew = fmaxf(mrow[j], mx);
      float a  = __expf(mrow[j] - mnew);
      float p0 = __expf(s0 - mnew);
      float p1 = __expf(s1 - mnew);
      float rs = p0 + p1;
      #pragma unroll
      for (int d = 1; d < 16; d <<= 1) rs += __shfl_xor(rs, d, 32);
      lrow[j] = lrow[j] * a + rs;
      mrow[j] = mnew;
      alpha[j] = a;
      Ps[wave][(hi * 8 + j) * LDP + lr]      = (half_t)p0;
      Ps[wave][(hi * 8 + j) * LDP + 16 + lr] = (half_t)p1;
    }
    #pragma unroll
    for (int g = 0; g < 4; ++g)
      #pragma unroll
      for (int j = 0; j < 8; ++j) O[g][j] *= alpha[j];

    // probs as A fragment (K = 32 keys)
    v16h ap;
    {
      const uint4* p = (const uint4*)(Ps[wave] + lr * LDP);
      ((uint4*)&ap)[0] = p[hi]; ((uint4*)&ap)[1] = p[2 + hi];
    }
    // O += P @ V  (four hd column groups)
    #pragma unroll
    for (int g = 0; g < 4; ++g) {
      const uint4* p = (const uint4*)(Vt + (g * 16 + lr) * LDV);
      v16h bf; ((uint4*)&bf)[0] = p[hi]; ((uint4*)&bf)[1] = p[2 + hi];
      O[g] = __builtin_amdgcn_wmma_f32_16x16x32_f16(
          false, ap, false, bf, (short)0, O[g], false, false);
    }
    __syncthreads();
  }

  // normalize and store context (f16), layout (B,S,H*HD)
  #pragma unroll
  for (int g = 0; g < 4; ++g)
    #pragma unroll
    for (int j = 0; j < 8; ++j) {
      float val = O[g][j] / lrow[j];
      size_t row = (size_t)(b * kS + q0 + hi * 8 + j);
      ctx[row * kD + hoff + g * 16 + lr] = (half_t)val;
    }
}

// ---------------------------------------------------------------- launcher
extern "C" void kernel_launch(void* const* d_in, const int* in_sizes, int n_in,
                              void* d_out, int out_size, void* d_ws, size_t ws_size,
                              hipStream_t stream) {
  const float* x     = (const float*)d_in[0];
  const float* ln1_g = (const float*)d_in[1];
  const float* ln1_b = (const float*)d_in[2];
  const float* ln2_g = (const float*)d_in[3];
  const float* ln2_b = (const float*)d_in[4];
  const float* wq    = (const float*)d_in[5];
  const float* bq    = (const float*)d_in[6];
  const float* wk    = (const float*)d_in[7];
  const float* bk    = (const float*)d_in[8];
  const float* wv    = (const float*)d_in[9];
  const float* bv    = (const float*)d_in[10];
  const float* wp    = (const float*)d_in[11];
  const float* bp    = (const float*)d_in[12];
  const float* ln3_g = (const float*)d_in[13];
  const float* ln3_b = (const float*)d_in[14];
  const float* w1    = (const float*)d_in[15];
  const float* b1    = (const float*)d_in[16];
  const float* w2    = (const float*)d_in[17];
  const float* b2    = (const float*)d_in[18];

  constexpr size_t DD  = (size_t)kD * kD;        // 1M
  constexpr size_t DFF = (size_t)kD * kFF;       // 4M
  constexpr size_t BSD = (size_t)kM * kD;        // 8M
  half_t* H = (half_t*)d_ws;
  half_t* wq16 = H;
  half_t* wk16 = wq16 + DD;
  half_t* wv16 = wk16 + DD;
  half_t* wp16 = wv16 + DD;
  half_t* w116 = wp16 + DD;
  half_t* w216 = w116 + DFF;
  half_t* xn16 = w216 + DFF;        // double-LN output
  half_t* q16  = xn16 + BSD;
  half_t* k16  = q16  + BSD;
  half_t* v16  = k16  + BSD;
  half_t* ctx16 = v16 + BSD;
  float*  x1   = (float*)(ctx16 + BSD);          // attn residual output (fp32)
  half_t* xn3  = (half_t*)(x1 + BSD);            // ln3 output
  half_t* h1   = xn16;                           // reuse xn/q/k/v region (64MB)

  // 1. weights -> f16 (one pass; they live in L2 afterwards)
  cvt_f16<<<(int)(DD / 1024), 256, 0, stream>>>(wq, wq16, (int)(DD / 4));
  cvt_f16<<<(int)(DD / 1024), 256, 0, stream>>>(wk, wk16, (int)(DD / 4));
  cvt_f16<<<(int)(DD / 1024), 256, 0, stream>>>(wv, wv16, (int)(DD / 4));
  cvt_f16<<<(int)(DD / 1024), 256, 0, stream>>>(wp, wp16, (int)(DD / 4));
  cvt_f16<<<(int)(DFF / 1024), 256, 0, stream>>>(w1, w116, (int)(DFF / 4));
  cvt_f16<<<(int)(DFF / 1024), 256, 0, stream>>>(w2, w216, (int)(DFF / 4));

  // 2. double pre-norm
  ln_kernel<<<kM, 256, 0, stream>>>(x, ln1_g, ln1_b, ln2_g, ln2_b, xn16, 1);

  // 3. QKV projections
  dim3 gP(kD / BN, kM / BM);
  gemm_wmma<0><<<gP, 128, 0, stream>>>(xn16, wq16, bq, nullptr, nullptr, q16, kM, kD, kD);
  gemm_wmma<0><<<gP, 128, 0, stream>>>(xn16, wk16, bk, nullptr, nullptr, k16, kM, kD, kD);
  gemm_wmma<0><<<gP, 128, 0, stream>>>(xn16, wv16, bv, nullptr, nullptr, v16, kM, kD, kD);

  // 4. fused flash attention
  attn_wmma<<<dim3(kS / 64, kH, kB), 128, 0, stream>>>(q16, k16, v16, ctx16);

  // 5. output projection + residual (fp32)
  gemm_wmma<2><<<gP, 128, 0, stream>>>(ctx16, wp16, bp, x, x1, nullptr, kM, kD, kD);

  // 6. ln3
  ln_kernel<<<kM, 256, 0, stream>>>(x1, ln3_g, ln3_b, ln3_g, ln3_b, xn3, 0);

  // 7. FFN up + exact GELU
  gemm_wmma<1><<<dim3(kFF / BN, kM / BM), 128, 0, stream>>>(
      xn3, w116, b1, nullptr, nullptr, h1, kM, kFF, kD);

  // 8. FFN down + residual -> final output (fp32)
  gemm_wmma<2><<<gP, 128, 0, stream>>>(h1, w216, b2, x1, (float*)d_out, nullptr,
                                       kM, kD, kFF);
}
